// RelationNetwork_60859686584940
// MI455X (gfx1250) — matrix operand order repaired and checked by reading, not compiled, not run on hardware
//
#include <hip/hip_runtime.h>

typedef __attribute__((ext_vector_type(16))) __bf16 v16bf;
typedef __attribute__((ext_vector_type(8)))  float  v8f;

#define BN_EPS 1e-5f

__device__ __forceinline__ unsigned short f32_bf16(float f) {
  union { float f; unsigned int u; } c; c.f = f;
  unsigned int u = c.u;
  u += 0x7fffu + ((u >> 16) & 1u);          // round-to-nearest-even
  return (unsigned short)(u >> 16);
}

union FragU { uint4 u[2]; v16bf v; };

// A-matrix 16x32 bf16 fragment per documented CDNA5 layout:
// lane: M = lane&15 ; halves 0..7 -> K = kc*32 + (lane>>4)*8 + 0..7 ; halves 8..15 -> +16
__device__ __forceinline__ v16bf loadA_lds(const unsigned short* base, int stride,
                                           int mt, int kc, int lane) {
  int m  = mt * 16 + (lane & 15);
  int kb = kc * 32 + ((lane >> 4) << 3);
  const unsigned short* p = base + m * stride + kb;
  FragU r;
  r.u[0] = *(const uint4*)(p);        // K = kb .. kb+7
  r.u[1] = *(const uint4*)(p + 16);   // K = kb+16 .. kb+23
  return r.v;
}

// B-matrix 32x16 bf16 fragment: lane: N = lane&15 ; halves 0..7 -> K = kc*32 + (lane>>4)*8 + 0..7 ; halves 8..15 -> +16
// Weights stored row-major [n][k] so each lane reads two 16-byte runs of its row, 32 bytes apart.
__device__ __forceinline__ v16bf loadB_glb(const unsigned short* __restrict__ W, int stride,
                                           int nt, int kc, int lane) {
  int n  = nt * 16 + (lane & 15);
  int kb = kc * 32 + ((lane >> 4) << 3);
  const unsigned short* p = W + n * stride + kb;
  FragU r;
  r.u[0] = *(const uint4*)(p);
  r.u[1] = *(const uint4*)(p + 16);
  return r.v;
}

// ---------------- conv + BN (stride 2, pad 1, 3x3) ----------------
__global__ void conv_bn_k(const float* __restrict__ in, const float* __restrict__ w,
                          const float* __restrict__ bias, const float* __restrict__ gam,
                          const float* __restrict__ bet, float* __restrict__ out,
                          int B, int CI, int CO, int H, int W) {
  int HO = H >> 1, WO = W >> 1;
  int total = B * CO * HO * WO;
  int idx = blockIdx.x * blockDim.x + threadIdx.x;
  if (idx >= total) return;
  int ox = idx % WO; int t = idx / WO;
  int oy = t % HO;   t /= HO;
  int co = t % CO;   int b = t / CO;
  float s = 0.f;
  const float* wp = w + co * CI * 9;
  for (int ci = 0; ci < CI; ++ci) {
    const float* ip = in + ((size_t)(b * CI + ci) * H) * W;
#pragma unroll
    for (int ky = 0; ky < 3; ++ky) {
      int iy = oy * 2 + ky - 1;
      if ((unsigned)iy >= (unsigned)H) continue;
#pragma unroll
      for (int kx = 0; kx < 3; ++kx) {
        int ix = ox * 2 + kx - 1;
        if ((unsigned)ix >= (unsigned)W) continue;
        s += ip[iy * W + ix] * wp[ci * 9 + ky * 3 + kx];
      }
    }
  }
  float scale = gam[co] * rsqrtf(1.f + BN_EPS);
  out[idx] = (s + bias[co]) * scale + bet[co];
}

// ---------------- build x_flat (B,25,26): 24 ch + 2 coords ----------------
__global__ void xflat_k(const float* __restrict__ c4, float* __restrict__ xf) {
  int idx = blockIdx.x * blockDim.x + threadIdx.x;
  if (idx >= 512 * 25 * 26) return;
  int ch = idx % 26; int t = idx / 26;
  int o = t % 25;    int b = t / 25;
  float v;
  if (ch < 24)       v = c4[(b * 24 + ch) * 25 + o];
  else if (ch == 24) v = (float)(o / 5 - 2) * 0.5f;
  else               v = (float)(o % 5 - 2) * 0.5f;
  xf[idx] = v;
}

// ---------------- convert g-weights fp32 -> bf16 ([n][k], K padded for g1) ----------------
__global__ void prep_w_k(const float* __restrict__ g1, const float* __restrict__ g2,
                         const float* __restrict__ g3, const float* __restrict__ g4,
                         unsigned short* __restrict__ w1p, unsigned short* __restrict__ w2p,
                         unsigned short* __restrict__ w3p, unsigned short* __restrict__ w4p) {
  int idx = blockIdx.x * blockDim.x + threadIdx.x;
  if (idx < 256 * 64) {
    int n = idx >> 6, k = idx & 63;
    w1p[idx] = (k < 63) ? f32_bf16(g1[n * 63 + k]) : (unsigned short)0;
    return;
  }
  int j = idx - 256 * 64;
  if (j < 3 * 65536) {
    const float* src = (j < 65536) ? g2 : (j < 131072 ? g3 : g4);
    unsigned short* dst = (j < 65536) ? w2p : (j < 131072 ? w3p : w4p);
    int r = j & 65535;
    dst[r] = f32_bf16(src[r]);
  }
}

// ---------------- fused g-MLP over pairs: 4 layers, WMMA bf16, pair sum ----------------
__global__ void __launch_bounds__(256, 1)
rn_g_wmma(const unsigned short* __restrict__ w1p, const unsigned short* __restrict__ w2p,
          const unsigned short* __restrict__ w3p, const unsigned short* __restrict__ w4p,
          const float* __restrict__ b1, const float* __restrict__ b2,
          const float* __restrict__ b3, const float* __restrict__ b4,
          const float* __restrict__ xf, const float* __restrict__ q,
          float* __restrict__ xg) {
  __shared__ float xfS[25 * 26];
  __shared__ float qS[16];
  __shared__ __align__(16) unsigned short pairS[64 * 64];    //  8 KB
  __shared__ __align__(16) unsigned short actS[64 * 256];    // 32 KB

  const int tid  = threadIdx.x;
  const int lane = tid & 31;
  const int w    = tid >> 5;          // wave 0..7 -> N-tiles {2w, 2w+1}
  const int b    = blockIdx.y;
  const int rowStart = blockIdx.x * 64;

  const v8f zero8 = {0.f, 0.f, 0.f, 0.f, 0.f, 0.f, 0.f, 0.f};

  // stage x_flat row block + q
  for (int i = tid; i < 650; i += 256) xfS[i] = xf[b * 650 + i];
  if (tid < 16) qS[tid] = (tid < 11) ? q[b * 11 + tid] : 0.f;
  __syncthreads();

  // build 64x64 bf16 pair tile: feat = [x_flat[c], x_flat[a], q, 0pad]
  {
    int r = tid >> 2;
    int quarter = tid & 3;
    int p = rowStart + r;
    bool valid = p < 625;
    int a = p / 25, c = p % 25;
#pragma unroll
    for (int cc = 0; cc < 16; ++cc) {
      int f = quarter * 16 + cc;
      float v = 0.f;
      if (valid) {
        if (f < 26)      v = xfS[c * 26 + f];
        else if (f < 52) v = xfS[a * 26 + (f - 26)];
        else if (f < 63) v = qS[f - 52];
      }
      pairS[r * 64 + f] = f32_bf16(v);
    }
  }
  __syncthreads();

  // ---- layer 1: pairS(64x64) x W1p(K=64) -> actS(64x256), relu
  // kc outer so each B fragment is loaded once and reused across the 4 M-tiles
  {
    float bias0 = b1[(w * 2 + 0) * 16 + (lane & 15)];
    float bias1 = b1[(w * 2 + 1) * 16 + (lane & 15)];
    v8f acc[4][2];
#pragma unroll
    for (int mt = 0; mt < 4; ++mt) { acc[mt][0] = zero8; acc[mt][1] = zero8; }
#pragma unroll
    for (int kc = 0; kc < 2; ++kc) {
      v16bf bb0 = loadB_glb(w1p, 64, w * 2 + 0, kc, lane);
      v16bf bb1 = loadB_glb(w1p, 64, w * 2 + 1, kc, lane);
#pragma unroll
      for (int mt = 0; mt < 4; ++mt) {
        v16bf a = loadA_lds(pairS, 64, mt, kc, lane);
        acc[mt][0] = __builtin_amdgcn_wmma_f32_16x16x32_bf16(false, a, false, bb0, (short)0, acc[mt][0], false, false);
        acc[mt][1] = __builtin_amdgcn_wmma_f32_16x16x32_bf16(false, a, false, bb1, (short)0, acc[mt][1], false, false);
      }
    }
#pragma unroll
    for (int mt = 0; mt < 4; ++mt)
#pragma unroll
      for (int v = 0; v < 8; ++v) {
        int m  = mt * 16 + v + ((lane >> 4) << 3);
        int n0 = (w * 2 + 0) * 16 + (lane & 15);
        int n1 = (w * 2 + 1) * 16 + (lane & 15);
        actS[m * 256 + n0] = f32_bf16(fmaxf(acc[mt][0][v] + bias0, 0.f));
        actS[m * 256 + n1] = f32_bf16(fmaxf(acc[mt][1][v] + bias1, 0.f));
      }
  }
  __syncthreads();

  // ---- layers 2..4: actS(64x256) x W(256x256); last layer reduces rows
  const unsigned short* Wl[3] = { w2p, w3p, w4p };
  const float*          Bl[3] = { b2,  b3,  b4  };
  float colsum0 = 0.f, colsum1 = 0.f;
  for (int l = 0; l < 3; ++l) {
    const unsigned short* Wc = Wl[l];
    float bias0 = Bl[l][(w * 2 + 0) * 16 + (lane & 15)];
    float bias1 = Bl[l][(w * 2 + 1) * 16 + (lane & 15)];
    v8f acc[4][2];
#pragma unroll
    for (int mt = 0; mt < 4; ++mt) { acc[mt][0] = zero8; acc[mt][1] = zero8; }
#pragma unroll
    for (int kc = 0; kc < 8; ++kc) {
      v16bf bb0 = loadB_glb(Wc, 256, w * 2 + 0, kc, lane);   // loaded once per kc,
      v16bf bb1 = loadB_glb(Wc, 256, w * 2 + 1, kc, lane);   // reused for 4 M-tiles
#pragma unroll
      for (int mt = 0; mt < 4; ++mt) {
        v16bf a = loadA_lds(actS, 256, mt, kc, lane);
        acc[mt][0] = __builtin_amdgcn_wmma_f32_16x16x32_bf16(false, a, false, bb0, (short)0, acc[mt][0], false, false);
        acc[mt][1] = __builtin_amdgcn_wmma_f32_16x16x32_bf16(false, a, false, bb1, (short)0, acc[mt][1], false, false);
      }
    }
    __syncthreads();   // all reads of actS complete before overwrite
    if (l < 2) {
#pragma unroll
      for (int mt = 0; mt < 4; ++mt)
#pragma unroll
        for (int v = 0; v < 8; ++v) {
          int m = mt * 16 + v + ((lane >> 4) << 3);
          actS[m * 256 + (w * 2 + 0) * 16 + (lane & 15)] = f32_bf16(fmaxf(acc[mt][0][v] + bias0, 0.f));
          actS[m * 256 + (w * 2 + 1) * 16 + (lane & 15)] = f32_bf16(fmaxf(acc[mt][1][v] + bias1, 0.f));
        }
      __syncthreads();
    } else {
#pragma unroll
      for (int mt = 0; mt < 4; ++mt)
#pragma unroll
        for (int v = 0; v < 8; ++v) {
          int p = rowStart + mt * 16 + v + ((lane >> 4) << 3);
          float h0 = fmaxf(acc[mt][0][v] + bias0, 0.f);
          float h1 = fmaxf(acc[mt][1][v] + bias1, 0.f);
          if (p < 625) { colsum0 += h0; colsum1 += h1; }
        }
    }
  }
  // combine the two 16-lane row-halves, then one atomic per column
  colsum0 += __shfl_xor(colsum0, 16, 32);
  colsum1 += __shfl_xor(colsum1, 16, 32);
  if (lane < 16) {
    atomicAdd(&xg[b * 256 + (w * 2 + 0) * 16 + lane], colsum0);
    atomicAdd(&xg[b * 256 + (w * 2 + 1) * 16 + lane], colsum1);
  }
}

// ---------------- f-MLP: 256 -> 256 -> 256 -> 10 ----------------
__global__ void __launch_bounds__(256)
rn_f(const float* __restrict__ xg,
     const float* __restrict__ f1w, const float* __restrict__ f1b,
     const float* __restrict__ f2w, const float* __restrict__ f2b,
     const float* __restrict__ f3w, const float* __restrict__ f3b,
     float* __restrict__ out) {
  __shared__ float xr[256], h1[256], h2[256];
  int b = blockIdx.x, t = threadIdx.x;
  xr[t] = xg[b * 256 + t];
  __syncthreads();
  {
    float s = f1b[t]; const float* wp = f1w + t * 256;
    for (int k = 0; k < 256; ++k) s += wp[k] * xr[k];
    h1[t] = fmaxf(s, 0.f);
  }
  __syncthreads();
  {
    float s = f2b[t]; const float* wp = f2w + t * 256;
    for (int k = 0; k < 256; ++k) s += wp[k] * h1[k];
    h2[t] = fmaxf(s, 0.f);
  }
  __syncthreads();
  if (t < 10) {
    float s = f3b[t]; const float* wp = f3w + t * 256;
    for (int k = 0; k < 256; ++k) s += wp[k] * h2[k];
    out[b * 10 + t] = s;
  }
}

extern "C" void kernel_launch(void* const* d_in, const int* in_sizes, int n_in,
                              void* d_out, int out_size, void* d_ws, size_t ws_size,
                              hipStream_t stream) {
  (void)in_sizes; (void)n_in; (void)out_size; (void)ws_size;
  const float* im = (const float*)d_in[0];
  const float* q  = (const float*)d_in[1];
  const float* cw[4] = {(const float*)d_in[2],  (const float*)d_in[6],
                        (const float*)d_in[10], (const float*)d_in[14]};
  const float* cb[4] = {(const float*)d_in[3],  (const float*)d_in[7],
                        (const float*)d_in[11], (const float*)d_in[15]};
  const float* bg[4] = {(const float*)d_in[4],  (const float*)d_in[8],
                        (const float*)d_in[12], (const float*)d_in[16]};
  const float* bb[4] = {(const float*)d_in[5],  (const float*)d_in[9],
                        (const float*)d_in[13], (const float*)d_in[17]};
  const float* g1w = (const float*)d_in[18]; const float* g1b = (const float*)d_in[19];
  const float* g2w = (const float*)d_in[20]; const float* g2b = (const float*)d_in[21];
  const float* g3w = (const float*)d_in[22]; const float* g3b = (const float*)d_in[23];
  const float* g4w = (const float*)d_in[24]; const float* g4b = (const float*)d_in[25];
  const float* f1w = (const float*)d_in[26]; const float* f1b = (const float*)d_in[27];
  const float* f2w = (const float*)d_in[28]; const float* f2b = (const float*)d_in[29];
  const float* f3w = (const float*)d_in[30]; const float* f3b = (const float*)d_in[31];
  float* out = (float*)d_out;

  char* ws = (char*)d_ws;
  size_t off = 0;
  auto wsa = [&](size_t bytes) { char* p = ws + off; off += (bytes + 511) & ~(size_t)511; return p; };
  float* c1 = (float*)wsa((size_t)512 * 24 * 40 * 40 * 4);
  float* c2 = (float*)wsa((size_t)512 * 24 * 20 * 20 * 4);
  float* c3 = (float*)wsa((size_t)512 * 24 * 10 * 10 * 4);
  float* c4 = (float*)wsa((size_t)512 * 24 * 5 * 5 * 4);
  float* xf = (float*)wsa((size_t)512 * 25 * 26 * 4);
  unsigned short* w1p = (unsigned short*)wsa((size_t)256 * 64 * 2);
  unsigned short* w2p = (unsigned short*)wsa((size_t)256 * 256 * 2);
  unsigned short* w3p = (unsigned short*)wsa((size_t)256 * 256 * 2);
  unsigned short* w4p = (unsigned short*)wsa((size_t)256 * 256 * 2);
  float* xg = (float*)wsa((size_t)512 * 256 * 4);

  // conv stack (memory-bound, negligible)
  {
    int tot = 512 * 24 * 40 * 40;
    conv_bn_k<<<(tot + 255) / 256, 256, 0, stream>>>(im, cw[0], cb[0], bg[0], bb[0], c1, 512, 3, 24, 80, 80);
    tot = 512 * 24 * 20 * 20;
    conv_bn_k<<<(tot + 255) / 256, 256, 0, stream>>>(c1, cw[1], cb[1], bg[1], bb[1], c2, 512, 24, 24, 40, 40);
    tot = 512 * 24 * 10 * 10;
    conv_bn_k<<<(tot + 255) / 256, 256, 0, stream>>>(c2, cw[2], cb[2], bg[2], bb[2], c3, 512, 24, 24, 20, 20);
    tot = 512 * 24 * 5 * 5;
    conv_bn_k<<<(tot + 255) / 256, 256, 0, stream>>>(c3, cw[3], cb[3], bg[3], bb[3], c4, 512, 24, 24, 10, 10);
  }

  xflat_k<<<(512 * 25 * 26 + 255) / 256, 256, 0, stream>>>(c4, xf);

  prep_w_k<<<(256 * 64 + 3 * 65536 + 255) / 256, 256, 0, stream>>>(g1w, g2w, g3w, g4w, w1p, w2p, w3p, w4p);

  hipMemsetAsync(xg, 0, (size_t)512 * 256 * 4, stream);

  // fused 4-layer g-MLP with WMMA: 5120 workgroups (10 row tiles x 512 batches)
  rn_g_wmma<<<dim3(10, 512), 256, 0, stream>>>(w1p, w2p, w3p, w4p,
                                               g1b, g2b, g3b, g4b, xf, q, xg);

  rn_f<<<512, 256, 0, stream>>>(xg, f1w, f1b, f2w, f2b, f3w, f3b, out);
}
